// BiodegradabilityPredictor_49795850830264
// MI455X (gfx1250) — matrix operand order repaired and checked by reading, not compiled, not run on hardware
//
#include <hip/hip_runtime.h>
#include <math.h>

// ---------------------------------------------------------------------------
// MI455X (gfx1250, wave32) MPNN forward.
// Heavy GEMMs via v_wmma_f32_16x16x32_f16 (f16 inputs, f32 accumulate).
// Weights staged in LDS (CDNA5: up to 320KB per workgroup). j-outer schedule
// (A resident in VGPRs, one accumulator tile at a time -> no spills) with
// one-step software pipelining of B ds_load_b128 pairs.
// ---------------------------------------------------------------------------

#define NN     50000
#define NE     800000
#define NG     1024
#define HDIM   128
#define NLAYER 4

typedef _Float16 f16;
typedef __attribute__((ext_vector_type(16))) _Float16 v16h;
typedef __attribute__((ext_vector_type(8)))  float    v8f;

__device__ __forceinline__ v8f wmma16(v16h a, v16h b, v8f c) {
  // D = A(16x32 f16) * B(32x16 f16) + C(16x16 f32)
  return __builtin_amdgcn_wmma_f32_16x16x32_f16(
      /*neg_a=*/false, a, /*neg_b=*/false, b,
      /*c_mod=*/(short)0, c, /*reuse_a=*/false, /*reuse_b=*/false);
}

// A-operand load from f32 row: lane holds K runs [koff..koff+7], [koff+16..koff+23]
__device__ __forceinline__ v16h ldA_f32(const float* __restrict__ p, int koff) {
  const float4* a = (const float4*)(p + koff);
  const float4* b = (const float4*)(p + koff + 16);
  float4 a0 = a[0], a1 = a[1], b0 = b[0], b1 = b[1];
  v16h r;
  r[0]=(f16)a0.x; r[1]=(f16)a0.y; r[2]=(f16)a0.z; r[3]=(f16)a0.w;
  r[4]=(f16)a1.x; r[5]=(f16)a1.y; r[6]=(f16)a1.z; r[7]=(f16)a1.w;
  r[8]=(f16)b0.x; r[9]=(f16)b0.y; r[10]=(f16)b0.z; r[11]=(f16)b0.w;
  r[12]=(f16)b1.x; r[13]=(f16)b1.y; r[14]=(f16)b1.z; r[15]=(f16)b1.w;
  return r;
}

// A-operand load from f16 row (LDS staging): same split-8 K layout
__device__ __forceinline__ v16h ldA_f16(const f16* p, int koff) {
  union { v16h v; uint4 q[2]; } u;
  u.q[0] = *(const uint4*)(p + koff);
  u.q[1] = *(const uint4*)(p + koff + 16);
  return u.v;
}

// B-operand load: lane holds K runs [kb..kb+7], [kb+16..kb+23] (split-8, matches A)
__device__ __forceinline__ v16h ldB_f16(const f16* p) {
  union { v16h v; uint4 q[2]; } u;
  u.q[0] = *(const uint4*)(p);
  u.q[1] = *(const uint4*)(p + 16);
  return u.v;
}

// ---------------------------------------------------------------------------
// Weight prep: src [L][K][N] f32  ->  dst [L][N][K] f16  (B-operand friendly)
// ---------------------------------------------------------------------------
__global__ void transpose_to_f16(const float* __restrict__ src,
                                 f16* __restrict__ dst,
                                 int tot, int K, int N) {
  int i = blockIdx.x * 256 + threadIdx.x;
  if (i >= tot) return;
  int k = i % K;
  int rem = i / K;
  int n = rem % N;
  int l = rem / N;
  dst[i] = (f16)src[((size_t)l * K + k) * N + n];
}

__global__ void zero_f32(float* __restrict__ p, int n) {
  int i = blockIdx.x * 256 + threadIdx.x;
  if (i < n) p[i] = 0.f;
}

// ---------------------------------------------------------------------------
// Node encoder: h = relu(x @ enc_W + enc_b), K=38 (tiny -> VALU)
// ---------------------------------------------------------------------------
__global__ void encoder_kernel(const float* __restrict__ x,
                               const float* __restrict__ W,
                               const float* __restrict__ b,
                               float* __restrict__ h) {
  int i = blockIdx.x * 256 + threadIdx.x;
  if (i >= NN * HDIM) return;
  int col = i & (HDIM - 1);
  int node = i >> 7;
  float acc = b[col];
  const float* xr = x + (size_t)node * 38;
  #pragma unroll
  for (int q = 0; q < 38; ++q) acc = fmaf(xr[q], W[q * HDIM + col], acc);
  h[i] = acc > 0.f ? acc : 0.f;
}

// ---------------------------------------------------------------------------
// Edge message kernel. One wave per 16-edge tile; 8 waves/block; each block
// handles EDGE_TPW*8 tiles so the one-time LDS weight preload is amortized.
// Dynamic LDS: W1T (96KB) + W2T (32KB) + 8x per-wave stage (32KB) = 160KB.
//   GEMM1: m1 = relu([h_dst || h_src || e] @ W1 + b1)   (K = 384)
//   GEMM2: m2 = m1 @ W2 + b2                            (K = 128)
//   scatter: atomicAdd into aggr[dst]
// ---------------------------------------------------------------------------
#define EDGE_TPW 5          // tiles per wave
#define EDGE_BLOCKS (NE / 16 / (8 * EDGE_TPW))   // 1250
#define W1_HALVES (128 * 384)
#define W2_HALVES (128 * 128)

__global__ __launch_bounds__(256)
void edge_msg_kernel(const float* __restrict__ h,
                     const int* __restrict__ srcI,
                     const int* __restrict__ dstI,
                     const float* __restrict__ eattr,
                     const float* __restrict__ eW,
                     const float* __restrict__ eb,
                     const f16* __restrict__ W1T,   // [128][384]
                     const float* __restrict__ b1,
                     const f16* __restrict__ W2T,   // [128][128]
                     const float* __restrict__ b2,
                     float* __restrict__ aggr) {
  extern __shared__ char smem[];
  f16* sW1   = (f16*)smem;                 // 98304 B
  f16* sW2   = sW1 + W1_HALVES;            // 32768 B
  f16* stage = sW2 + W2_HALVES;            // 8 * 4096 B

  const int tid  = threadIdx.x;
  const int wave = tid >> 5;
  const int lane = tid & 31;

  // --- cooperative weight preload (uint4 = 8 halves) ---
  {
    const uint4* g1 = (const uint4*)W1T;
    uint4* s1 = (uint4*)sW1;
    #pragma unroll
    for (int i = 0; i < W1_HALVES / 8 / 256; ++i) s1[tid + i * 256] = g1[tid + i * 256];
    const uint4* g2 = (const uint4*)W2T;
    uint4* s2 = (uint4*)sW2;
    #pragma unroll
    for (int i = 0; i < W2_HALVES / 8 / 256; ++i) s2[tid + i * 256] = g2[tid + i * 256];
  }
  __syncthreads();

  const int arow  = lane & 15;              // A row this lane feeds
  const int koff  = (lane >> 4) * 8;        // A K sub-offset
  const int kb    = (lane >> 4) * 8;        // B K sub-offset
  const int bcol  = lane & 15;              // B/D column
  const int rhalf = (lane >> 4) * 8;        // D row half
  f16* myStage = stage + wave * (16 * HDIM);

  #pragma clang loop unroll(disable)
  for (int t = 0; t < EDGE_TPW; ++t) {
    const int tile = blockIdx.x * (8 * EDGE_TPW) + t * 8 + wave;
    const int ebase = tile * 16;
    const int eA = ebase + arow;
    const int s  = srcI[eA];
    const int d2 = dstI[eA];
    const float* hd = h + (size_t)d2 * HDIM;
    const float* hs = h + (size_t)s  * HDIM;

    // ---- gather A: h[dst] | h[src] | e(edge MLP on the fly), f16 A layout
    v16h A[12];
    #pragma unroll
    for (int c = 0; c < 4; ++c) A[c]     = ldA_f32(hd + c * 32, koff);
    #pragma unroll
    for (int c = 0; c < 4; ++c) A[4 + c] = ldA_f32(hs + c * 32, koff);

    float ea[7];
    #pragma unroll
    for (int q = 0; q < 7; ++q) ea[q] = eattr[(size_t)eA * 7 + q];
    #pragma unroll
    for (int c = 0; c < 4; ++c) {
      v16h tv;
      #pragma unroll
      for (int hh = 0; hh < 16; ++hh) {
        int col = c * 32 + koff + hh + ((hh >= 8) ? 8 : 0);
        float a = eb[col];
        #pragma unroll
        for (int q = 0; q < 7; ++q) a = fmaf(ea[q], eW[q * HDIM + col], a);
        tv[hh] = (f16)(a > 0.f ? a : 0.f);
      }
      A[8 + c] = tv;
    }

    // ---- GEMM1: j-outer (A resident, single acc tile), B pipelined 1 step
    #pragma unroll
    for (int j = 0; j < 8; ++j) {
      const float bv = b1[j * 16 + bcol];
      v8f acc;
      #pragma unroll
      for (int u = 0; u < 8; ++u) acc[u] = bv;
      const f16* bp = sW1 + (size_t)(j * 16 + bcol) * 384 + kb;
      v16h bc = ldB_f16(bp);
      #pragma unroll
      for (int c = 0; c < 12; ++c) {
        v16h bn = (c < 11) ? ldB_f16(bp + (c + 1) * 32) : bc;
        acc = wmma16(A[c], bc, acc);
        bc = bn;
      }
      #pragma unroll
      for (int u = 0; u < 8; ++u) {
        float v = acc[u] > 0.f ? acc[u] : 0.f;
        myStage[(rhalf + u) * HDIM + j * 16 + bcol] = (f16)v;
      }
    }
    __syncthreads();

    // ---- GEMM2: staged m1 (A layout) @ W2, K=128, same schedule
    v16h A2[4];
    const f16* sp = myStage + arow * HDIM;
    #pragma unroll
    for (int c = 0; c < 4; ++c) A2[c] = ldA_f16(sp + c * 32, koff);

    int nd[8];
    #pragma unroll
    for (int u = 0; u < 8; ++u) nd[u] = dstI[ebase + rhalf + u];

    #pragma unroll
    for (int j = 0; j < 8; ++j) {
      const float bv = b2[j * 16 + bcol];
      v8f acc;
      #pragma unroll
      for (int u = 0; u < 8; ++u) acc[u] = bv;
      const f16* bp = sW2 + (size_t)(j * 16 + bcol) * HDIM + kb;
      v16h bc = ldB_f16(bp);
      #pragma unroll
      for (int c = 0; c < 4; ++c) {
        v16h bn = (c < 3) ? ldB_f16(bp + (c + 1) * 32) : bc;
        acc = wmma16(A2[c], bc, acc);
        bc = bn;
      }
      #pragma unroll
      for (int u = 0; u < 8; ++u)
        atomicAdd(aggr + (size_t)nd[u] * HDIM + j * 16 + bcol, acc[u]);
    }
    __syncthreads();   // protect stage before next iteration overwrites
  }
}

// ---------------------------------------------------------------------------
// Generic node GEMM: C[M][Nout] = A[M][128](f32) @ WT[Nout][128](f16)^T + bias
// WT preloaded to LDS once per block; grid-stride loop over 16-row tiles.
// Dynamic LDS = Nout*128*2 bytes (max 96KB for Nout=384).
// ---------------------------------------------------------------------------
__global__ __launch_bounds__(256)
void gemm_k128(const float* __restrict__ A,
               const f16* __restrict__ WT,
               const float* __restrict__ bias,
               float* __restrict__ C,
               int Mtiles, int Nout) {
  extern __shared__ char smem[];
  f16* sW = (f16*)smem;
  const int tid  = threadIdx.x;
  const int wave = tid >> 5;
  const int lane = tid & 31;

  {
    const int nq = (Nout * HDIM) / 8;   // uint4 count
    const uint4* g = (const uint4*)WT;
    uint4* s = (uint4*)sW;
    for (int i = tid; i < nq; i += 256) s[i] = g[i];
  }
  __syncthreads();

  const int koff  = (lane >> 4) * 8;
  const int kb    = (lane >> 4) * 8;
  const int bcol  = lane & 15;
  const int rhalf = (lane >> 4) * 8;
  const int nt = Nout >> 4;

  #pragma clang loop unroll(disable)
  for (int tile = blockIdx.x * 8 + wave; tile < Mtiles; tile += gridDim.x * 8) {
    const int rbase = tile * 16;
    const float* arow = A + (size_t)(rbase + (lane & 15)) * HDIM;
    v16h Ar[4];
    #pragma unroll
    for (int c = 0; c < 4; ++c) Ar[c] = ldA_f32(arow + c * 32, koff);

    #pragma clang loop unroll(disable)
    for (int j = 0; j < nt; ++j) {
      const float bv = bias[j * 16 + bcol];
      v8f acc;
      #pragma unroll
      for (int u = 0; u < 8; ++u) acc[u] = bv;
      const f16* bp = sW + (size_t)(j * 16 + bcol) * HDIM + kb;
      v16h bc = ldB_f16(bp);
      #pragma unroll
      for (int c = 0; c < 4; ++c) {
        v16h bn = (c < 3) ? ldB_f16(bp + (c + 1) * 32) : bc;
        acc = wmma16(Ar[c], bc, acc);
        bc = bn;
      }
      #pragma unroll
      for (int u = 0; u < 8; ++u)
        C[(size_t)(rbase + rhalf + u) * Nout + j * 16 + bcol] = acc[u];
    }
  }
}

// ---------------------------------------------------------------------------
// GRU + BatchNorm(eval) + relu + skip add (elementwise)
// ---------------------------------------------------------------------------
__global__ void gru_kernel(const float* __restrict__ gi,
                           const float* __restrict__ gh,
                           const float* __restrict__ h,
                           const float* __restrict__ skp,
                           const float* __restrict__ gma,
                           const float* __restrict__ bta,
                           const float* __restrict__ mea,
                           const float* __restrict__ var,
                           float* __restrict__ hout) {
  int i = blockIdx.x * 256 + threadIdx.x;
  if (i >= NN * HDIM) return;
  int col = i & (HDIM - 1);
  size_t node = (size_t)(i >> 7);
  size_t o = node * 384 + col;
  float ir = gi[o], iz = gi[o + 128], in_ = gi[o + 256];
  float hr = gh[o], hz = gh[o + 128], hn_ = gh[o + 256];
  float hc = h[i];
  float r = 1.f / (1.f + expf(-(ir + hr)));
  float z = 1.f / (1.f + expf(-(iz + hz)));
  float nn = tanhf(in_ + r * hn_);
  float hv = (1.f - z) * nn + z * hc;
  hv = (hv - mea[col]) * rsqrtf(var[col] + 1e-5f) * gma[col] + bta[col];
  hout[i] = (hv > 0.f ? hv : 0.f) + skp[i];
}

// ---------------------------------------------------------------------------
// Pooling: segment mean + max over graphs (order-preserving uint max encoding)
// ---------------------------------------------------------------------------
__device__ __forceinline__ unsigned enc_max(float f) {
  unsigned u = __float_as_uint(f);
  return (u >> 31) ? ~u : (u | 0x80000000u);
}
__device__ __forceinline__ float dec_max(unsigned k) {
  unsigned u = (k & 0x80000000u) ? (k & 0x7FFFFFFFu) : ~k;
  return __uint_as_float(u);
}

__global__ void pool_zero(float* __restrict__ sum, unsigned* __restrict__ mx,
                          float* __restrict__ cnt) {
  int i = blockIdx.x * 256 + threadIdx.x;
  if (i < NG * HDIM) { sum[i] = 0.f; mx[i] = 0x007FFFFFu; /* enc(-inf) */ }
  if (i < NG) cnt[i] = 0.f;
}

__global__ void pool_scatter(const float* __restrict__ h,
                             const int* __restrict__ batch,
                             float* __restrict__ sum,
                             unsigned* __restrict__ mx,
                             float* __restrict__ cnt) {
  int i = blockIdx.x * 256 + threadIdx.x;
  if (i >= NN * HDIM) return;
  int col = i & (HDIM - 1);
  int node = i >> 7;
  int g = batch[node];
  float v = h[i];
  atomicAdd(&sum[(size_t)g * HDIM + col], v);
  atomicMax(&mx[(size_t)g * HDIM + col], enc_max(v));
  if (col == 0) atomicAdd(&cnt[g], 1.0f);
}

__global__ void pool_finalize(const float* __restrict__ sum,
                              const unsigned* __restrict__ mx,
                              const float* __restrict__ cnt,
                              float* __restrict__ g) {
  int i = blockIdx.x * 256 + threadIdx.x;
  if (i >= NG * 256) return;
  int c = i & 255;
  int gr = i >> 8;
  if (c < HDIM)
    g[i] = sum[(size_t)gr * HDIM + c] / fmaxf(cnt[gr], 1.0f);
  else
    g[i] = dec_max(mx[(size_t)gr * HDIM + (c - HDIM)]);
}

// ---------------------------------------------------------------------------
// Readout MLP (tiny -> VALU)
// ---------------------------------------------------------------------------
__global__ void readout1(const float* __restrict__ g, const float* __restrict__ W,
                         const float* __restrict__ b, float* __restrict__ o) {
  int i = blockIdx.x * 256 + threadIdx.x;
  if (i >= NG * 128) return;
  int c = i & 127, gr = i >> 7;
  float acc = b[c];
  const float* gp = g + (size_t)gr * 256;
  #pragma unroll 8
  for (int q = 0; q < 256; ++q) acc = fmaf(gp[q], W[q * 128 + c], acc);
  o[i] = acc > 0.f ? acc : 0.f;
}
__global__ void readout2(const float* __restrict__ a, const float* __restrict__ W,
                         const float* __restrict__ b, float* __restrict__ o) {
  int i = blockIdx.x * 256 + threadIdx.x;
  if (i >= NG * 64) return;
  int c = i & 63, gr = i >> 6;
  float acc = b[c];
  const float* ap = a + (size_t)gr * 128;
  #pragma unroll 8
  for (int q = 0; q < 128; ++q) acc = fmaf(ap[q], W[q * 64 + c], acc);
  o[i] = acc > 0.f ? acc : 0.f;
}
__global__ void readout3(const float* __restrict__ a, const float* __restrict__ W,
                         const float* __restrict__ b, float* __restrict__ out) {
  int i = blockIdx.x * 256 + threadIdx.x;
  if (i >= NG) return;
  float acc = b[0];
  const float* ap = a + (size_t)i * 64;
  #pragma unroll 8
  for (int q = 0; q < 64; ++q) acc = fmaf(ap[q], W[q], acc);
  out[i] = 1.f / (1.f + expf(-acc));
}

// ---------------------------------------------------------------------------
// Host launcher. Workspace budget ~260 MB.
// ---------------------------------------------------------------------------
extern "C" void kernel_launch(void* const* d_in, const int* in_sizes, int n_in,
                              void* d_out, int out_size, void* d_ws, size_t ws_size,
                              hipStream_t stream) {
  (void)in_sizes; (void)n_in; (void)out_size; (void)ws_size;
  const float* x      = (const float*)d_in[0];
  const float* eattr  = (const float*)d_in[1];
  const int*   eidx   = (const int*)  d_in[2];
  const int*   batch  = (const int*)  d_in[3];
  const float* enc_W  = (const float*)d_in[4];
  const float* enc_b  = (const float*)d_in[5];
  const float* edge_W = (const float*)d_in[6];
  const float* edge_b = (const float*)d_in[7];
  const float* msg_W1 = (const float*)d_in[8];
  const float* msg_b1 = (const float*)d_in[9];
  const float* msg_W2 = (const float*)d_in[10];
  const float* msg_b2 = (const float*)d_in[11];
  const float* gWih   = (const float*)d_in[12];
  const float* gbih   = (const float*)d_in[13];
  const float* gWhh   = (const float*)d_in[14];
  const float* gbhh   = (const float*)d_in[15];
  const float* bn_g   = (const float*)d_in[16];
  const float* bn_b   = (const float*)d_in[17];
  const float* bn_m   = (const float*)d_in[18];
  const float* bn_v   = (const float*)d_in[19];
  const float* skW    = (const float*)d_in[20];
  const float* skb    = (const float*)d_in[21];
  const float* r1W    = (const float*)d_in[22];
  const float* r1b    = (const float*)d_in[23];
  const float* r2W    = (const float*)d_in[24];
  const float* r2b    = (const float*)d_in[25];
  const float* r3W    = (const float*)d_in[26];
  const float* r3b    = (const float*)d_in[27];
  const int* srcI = eidx;            // edge_index[0]
  const int* dstI = eidx + NE;       // edge_index[1]

  char* base = (char*)d_ws;
  size_t off = 0;
  auto alloc = [&](size_t bytes) -> char* {
    char* p = base + off;
    off = (off + bytes + 255) & ~(size_t)255;
    return p;
  };
  float*    hA   = (float*)   alloc((size_t)NN * HDIM * 4);
  float*    hB   = (float*)   alloc((size_t)NN * HDIM * 4);
  float*    aggr = (float*)   alloc((size_t)NN * HDIM * 4);
  float*    gi   = (float*)   alloc((size_t)NN * 384 * 4);
  float*    gh   = (float*)   alloc((size_t)NN * 384 * 4);
  float*    skp  = (float*)   alloc((size_t)NN * HDIM * 4);
  float*    psum = (float*)   alloc((size_t)NG * HDIM * 4);
  unsigned* pmax = (unsigned*)alloc((size_t)NG * HDIM * 4);
  float*    pcnt = (float*)   alloc((size_t)NG * 4);
  float*    g    = (float*)   alloc((size_t)NG * 256 * 4);
  float*    ro1  = (float*)   alloc((size_t)NG * HDIM * 4);
  float*    ro2  = (float*)   alloc((size_t)NG * 64 * 4);
  f16*      W1T  = (f16*)     alloc((size_t)NLAYER * W1_HALVES * 2);
  f16*      W2T  = (f16*)     alloc((size_t)NLAYER * W2_HALVES * 2);
  f16*      WihT = (f16*)     alloc((size_t)NLAYER * 384 * 128 * 2);
  f16*      WhhT = (f16*)     alloc((size_t)NLAYER * 384 * 128 * 2);
  f16*      SkT  = (f16*)     alloc((size_t)NLAYER * 128 * 128 * 2);

  // ---- weight prep: f32 [L][K][N] -> f16 [L][N][K]
  auto tlaunch = [&](const float* s, f16* d, int K, int N) {
    int tot = NLAYER * K * N;
    transpose_to_f16<<<(tot + 255) / 256, 256, 0, stream>>>(s, d, tot, K, N);
  };
  tlaunch(msg_W1, W1T, 384, 128);
  tlaunch(msg_W2, W2T, 128, 128);
  tlaunch(gWih,  WihT, 128, 384);
  tlaunch(gWhh,  WhhT, 128, 384);
  tlaunch(skW,   SkT,  128, 128);

  // ---- node encoder
  encoder_kernel<<<(NN * HDIM + 255) / 256, 256, 0, stream>>>(x, enc_W, enc_b, hA);

  const int MT = NN / 16;                    // 3125 row tiles (NN % 16 == 0)
  const size_t edge_lds = (size_t)(W1_HALVES + W2_HALVES + 8 * 16 * HDIM) * 2; // 160KB
  const size_t g384_lds = (size_t)384 * HDIM * 2;   // 96KB
  const size_t g128_lds = (size_t)128 * HDIM * 2;   // 32KB

  float* hc = hA;
  float* hn = hB;
  for (int l = 0; l < NLAYER; ++l) {
    zero_f32<<<(NN * HDIM + 255) / 256, 256, 0, stream>>>(aggr, NN * HDIM);
    edge_msg_kernel<<<EDGE_BLOCKS, 256, edge_lds, stream>>>(
        hc, srcI, dstI, eattr, edge_W, edge_b,
        W1T + (size_t)l * W1_HALVES, msg_b1 + l * 128,
        W2T + (size_t)l * W2_HALVES, msg_b2 + l * 128, aggr);
    gemm_k128<<<160, 256, g384_lds, stream>>>(
        aggr, WihT + (size_t)l * 384 * 128, gbih + l * 384, gi, MT, 384);
    gemm_k128<<<160, 256, g384_lds, stream>>>(
        hc, WhhT + (size_t)l * 384 * 128, gbhh + l * 384, gh, MT, 384);
    gemm_k128<<<160, 256, g128_lds, stream>>>(
        hc, SkT + (size_t)l * 128 * 128, skb + l * 128, skp, MT, 128);
    gru_kernel<<<(NN * HDIM + 255) / 256, 256, 0, stream>>>(
        gi, gh, hc, skp, bn_g + l * 128, bn_b + l * 128,
        bn_m + l * 128, bn_v + l * 128, hn);
    float* t = hc; hc = hn; hn = t;
  }

  // ---- pooling + readout
  pool_zero<<<(NG * HDIM + 255) / 256, 256, 0, stream>>>(psum, pmax, pcnt);
  pool_scatter<<<(NN * HDIM + 255) / 256, 256, 0, stream>>>(hc, batch, psum, pmax, pcnt);
  pool_finalize<<<(NG * 256 + 255) / 256, 256, 0, stream>>>(psum, pmax, pcnt, g);
  readout1<<<(NG * 128 + 255) / 256, 256, 0, stream>>>(g, r1W, r1b, ro1);
  readout2<<<(NG * 64 + 255) / 256, 256, 0, stream>>>(ro1, r2W, r2b, ro2);
  readout3<<<(NG + 255) / 256, 256, 0, stream>>>(ro2, r3W, r3b, (float*)d_out);
}
